// ISA_Block_CA_88493506167126
// MI455X (gfx1250) — matrix operand order repaired and checked by reading, not compiled, not run on hardware
//
#include <hip/hip_runtime.h>

// ---------------------------------------------------------------------------
// MI455X (gfx1250) block cross-attention pipeline, bf16 WMMA throughout,
// with GLOBAL_LOAD_ASYNC_TO_LDS staging (ASYNCcnt) where available.
// Long stage:  L=64,  B=512, C=128, 8 heads, dh=16
// Short stage: L=1024, B=32, C=128, flash-style streaming attention
// ---------------------------------------------------------------------------

typedef __bf16 bf16;
typedef __attribute__((ext_vector_type(16))) __bf16 v16bf;
typedef __attribute__((ext_vector_type(8)))  float  v8f;
typedef __attribute__((ext_vector_type(4)))  int    v4i;

#define CDIM   128
#define TDIM   4
#define HDIM   64
#define WDIM   64
#define NHEAD  8
#define DH     16
#define MTOK   32768          // total tokens (both stages)

// ---- async global->LDS copy (gfx1250 GLOBAL_LOAD_ASYNC_TO_LDS_B128) -------
#if defined(__has_builtin)
#if __has_builtin(__builtin_amdgcn_global_load_async_to_lds_b128) && \
    __has_builtin(__builtin_amdgcn_s_wait_asynccnt)
#define USE_ASYNC_LDS 1
#endif
#endif
#ifndef USE_ASYNC_LDS
#define USE_ASYNC_LDS 0
#endif

__device__ inline void cp_async_b128(const void* g, void* l) {
#if USE_ASYNC_LDS
  // Builtin signature (from hipcc diagnostic): (int4 AS1*, int4 AS3*, Imm, Imm).
  // Global generic pointer value == 64-bit VA; LDS generic pointer low 32
  // bits == LDS offset (matches LLVM flat->local addrspacecast lowering).
  __builtin_amdgcn_global_load_async_to_lds_b128(
      (__attribute__((address_space(1))) v4i*)(unsigned long long)g,
      (__attribute__((address_space(3))) v4i*)(unsigned int)(unsigned long long)l,
      0, 0);
#else
  *(uint4*)l = *(const uint4*)g;
#endif
}

__device__ inline void cp_async_wait() {
#if USE_ASYNC_LDS
  __builtin_amdgcn_s_wait_asynccnt(0);
#endif
}

// ---- WMMA helpers ---------------------------------------------------------
__device__ inline v8f wmma_bf16(v16bf a, v16bf b, v8f c) {
  return __builtin_amdgcn_wmma_f32_16x16x32_bf16(false, a, false, b, (short)0, c,
                                                 false, false);
}

// A-matrix 16x32 bf16 fragment from row-major [m][k] (stride in elements).
// Contiguous k order used here, not the ISA table order: lane = half*16 + m%16 ; VGPR v: K = half*16 + v*2
__device__ inline v16bf frag_a(const bf16* base, int stride, int m, int half, int k0) {
  v16bf a;
#pragma unroll
  for (int v = 0; v < 8; ++v) {
    int kb = k0 + (half << 4) + (v << 1);
    const bf16* p = base + m * stride + kb;
    a[2 * v]     = p[0];
    a[2 * v + 1] = p[1];
  }
  return a;
}

// B-matrix 32x16 fragment where B[k][n] is sourced from row-major S[n][k].
__device__ inline v16bf frag_b_nk(const bf16* base, int stride, int n, int half, int k0) {
  v16bf b;
#pragma unroll
  for (int v = 0; v < 8; ++v) {
    int k = k0 + (half << 4) + (v << 1);
    const bf16* p = base + n * stride + k;
    b[2 * v]     = p[0];
    b[2 * v + 1] = p[1];
  }
  return b;
}

// B-matrix fragment from row-major S[k][n] (e.g. V tile: key-major).
__device__ inline v16bf frag_b_kn(const bf16* base, int stride, int n, int half, int k0) {
  v16bf b;
#pragma unroll
  for (int v = 0; v < 8; ++v) {
    int k = k0 + (half << 4) + (v << 1);
    b[2 * v]     = base[k * stride + n];
    b[2 * v + 1] = base[(k + 1) * stride + n];
  }
  return b;
}

// ---------------------------------------------------------------------------
// Kernel 1: NCTHW -> long-token-major bf16 with positional add.
// long token: l = (t*4+bnh)*4+bnw (0..63), b = (n*16+hb)*16+wb (0..511)
// ---------------------------------------------------------------------------
__global__ __launch_bounds__(256) void pack_long_kernel(
    const float* __restrict__ q, const float* __restrict__ k,
    const float* __restrict__ v, const float* __restrict__ pq,
    const float* __restrict__ pk, bf16* __restrict__ Xq,
    bf16* __restrict__ Xk, bf16* __restrict__ Xv) {
  int id = blockIdx.x * 256 + threadIdx.x;   // input-major (n,c,t,h,w)
  int w   = id & 63;
  int hh  = (id >> 6) & 63;
  int t   = (id >> 12) & 3;
  int c   = (id >> 14) & 127;
  int n   = id >> 21;
  int bnh = hh >> 4, hb = hh & 15, bnw = w >> 4, wb = w & 15;
  int l  = (t * 4 + bnh) * 4 + bnw;
  int bb = (n * 16 + hb) * 16 + wb;
  size_t o = ((size_t)(l * 512 + bb)) * CDIM + c;
  Xq[o] = (bf16)(q[id] + pq[id]);
  Xk[o] = (bf16)(k[id] + pk[id]);
  Xv[o] = (bf16)v[id];
}

// ---------------------------------------------------------------------------
// Kernel 2: tensor-core GEMM  Y = (X @ W^T + bias) * scale
// 64 rows x full 128 cols, K=128 in LDS; 8 waves x 4 tiles x 4 k-steps.
// MODE 0: bf16 token-major | 1: fp32 token-major | 2: fp32 NCTHW scatter
// ---------------------------------------------------------------------------
template <int MODE>
__global__ __launch_bounds__(256) void gemm_tc_kernel(
    const bf16* __restrict__ X0, const bf16* __restrict__ X1,
    const bf16* __restrict__ X2, const float* __restrict__ Wt,
    const float* __restrict__ bias, bf16* __restrict__ Y0,
    bf16* __restrict__ Y1, bf16* __restrict__ Y2, float* __restrict__ Yf,
    float scale0) {
  const int z = blockIdx.z;
  const bf16* X = (z == 0) ? X0 : (z == 1) ? X1 : X2;
  bf16* Yb      = (z == 0) ? Y0 : (z == 1) ? Y1 : Y2;
  const float* Wz = Wt + (size_t)z * CDIM * CDIM;
  const float* bz = bias + z * CDIM;
  const float scale = (z == 0) ? scale0 : 1.0f;
  const int m0 = blockIdx.x * 64;

  __shared__ alignas(16) bf16 As[64][CDIM];     // 16 KB
  __shared__ alignas(16) bf16 Ws[CDIM][CDIM];   // 32 KB, row-major (n,k)

  const int tid = threadIdx.x;
  // Stage A tile via async DMA: 64*128 bf16 = 1024 x 16B chunks.
  {
    const char* g = (const char*)(X + (size_t)m0 * CDIM);
    char* l = (char*)&As[0][0];
    for (int i = tid; i < 1024; i += 256)
      cp_async_b128(g + i * 16, l + i * 16);
  }
  // Stage weights with fp32 -> bf16 conversion (VALU path).
  {
    bf16* dw = &Ws[0][0];
    for (int i = tid; i < CDIM * CDIM; i += 256) dw[i] = (bf16)Wz[i];
  }
  cp_async_wait();
  __syncthreads();

  const int wid = tid >> 5, lane = tid & 31, half = lane >> 4;
  const int mstrip = wid >> 1;            // 0..3 (16-row strip)
  const int ntbase = (wid & 1) * 4;       // 4 n-tiles per wave
  const int mrow = mstrip * 16 + (lane & 15);

  v16bf afr[4];
#pragma unroll
  for (int kk = 0; kk < 4; ++kk)
    afr[kk] = frag_a(&As[0][0], CDIM, mrow, half, kk * 32);

  for (int nt = 0; nt < 4; ++nt) {
    const int n0 = (ntbase + nt) * 16;
    const int ncol = n0 + (lane & 15);
    v8f acc = {};
#pragma unroll
    for (int kk = 0; kk < 4; ++kk) {
      v16bf bfr = frag_b_nk(&Ws[0][0], CDIM, ncol, half, kk * 32);
      acc = wmma_bf16(afr[kk], bfr, acc);
    }
    const float bv = bz[ncol];
#pragma unroll
    for (int r = 0; r < 8; ++r) {
      const int m = m0 + mstrip * 16 + r + half * 8;   // D row = r + 8*half
      const float val = (acc[r] + bv) * scale;
      if (MODE == 0) {
        Yb[(size_t)m * CDIM + ncol] = (bf16)val;
      } else if (MODE == 1) {
        Yf[(size_t)m * CDIM + ncol] = val;
      } else {
        // short token m = l'*32 + b' ; l' = (t,hb,wb), b' = (n,bnh,bnw)
        const int bb = m & 31, lp = m >> 5;
        const int t = lp >> 8, hb = (lp >> 4) & 15, wb2 = lp & 15;
        const int nn = bb >> 4, bnh = (bb >> 2) & 3, bnw = bb & 3;
        const size_t oidx =
            ((((size_t)nn * CDIM + ncol) * TDIM + t) * HDIM + (bnh * 16 + hb)) *
                WDIM + (bnw * 16 + wb2);
        Yf[oidx] = val;
      }
    }
  }
}

// ---------------------------------------------------------------------------
// Kernel 3: long attention. One block per (b,h); L=64 keys fit in LDS.
// ---------------------------------------------------------------------------
__global__ __launch_bounds__(128) void attn_long_kernel(
    const bf16* __restrict__ QP, const bf16* __restrict__ KP,
    const bf16* __restrict__ VP, bf16* __restrict__ AO) {
  const int bh = blockIdx.x, b = bh >> 3, h = bh & 7;
  __shared__ alignas(16) bf16 Qs[64][32];     // K-dim zero-padded to 32
  __shared__ alignas(16) bf16 Ks[64][32];
  __shared__ alignas(16) bf16 Vv[64][DH];
  __shared__ alignas(16) float Sf[64][64];
  __shared__ alignas(16) bf16 Pm[64][64];

  const int tid = threadIdx.x;
  {
    // 128 threads cover 64 rows x two 16B chunks (32B = 16 bf16 per row).
    const int r = tid >> 1, hs = tid & 1;
    const size_t gb = ((size_t)(r * 512 + b)) * CDIM + h * DH + hs * 8;
    cp_async_b128(QP + gb, &Qs[r][hs * 8]);
    cp_async_b128(KP + gb, &Ks[r][hs * 8]);
    cp_async_b128(VP + gb, &Vv[r][hs * 8]);
    const uint4 z4 = make_uint4(0u, 0u, 0u, 0u);
    *(uint4*)&Qs[r][16 + hs * 8] = z4;   // zero the K-dim pad (cols 16..31)
    *(uint4*)&Ks[r][16 + hs * 8] = z4;
  }
  cp_async_wait();
  __syncthreads();

  const int wid = tid >> 5, lane = tid & 31, half = lane >> 4;
  const int mrow = wid * 16 + (lane & 15);
  const v16bf aq = frag_a(&Qs[0][0], 32, mrow, half, 0);

  // S = Q K^T  (dh padded to 32 -> one WMMA per 16x16 tile)
  for (int nt = 0; nt < 4; ++nt) {
    const int ncol = nt * 16 + (lane & 15);
    v16bf bk = frag_b_nk(&Ks[0][0], 32, ncol, half, 0);
    v8f s = {};
    s = wmma_bf16(aq, bk, s);
#pragma unroll
    for (int r = 0; r < 8; ++r) Sf[wid * 16 + r + half * 8][ncol] = s[r];
  }
  __syncthreads();

  if (tid < 64) {
    float mx = -3.0e38f;
    for (int j = 0; j < 64; ++j) mx = fmaxf(mx, Sf[tid][j]);
    float sum = 0.0f;
    for (int j = 0; j < 64; ++j) {
      const float p = __expf(Sf[tid][j] - mx);
      Sf[tid][j] = p;
      sum += p;
    }
    const float inv = 1.0f / sum;
    for (int j = 0; j < 64; ++j) Pm[tid][j] = (bf16)(Sf[tid][j] * inv);
  }
  __syncthreads();

  // O = P V  (K=64 -> two WMMA k-steps)
  v8f o = {};
#pragma unroll
  for (int kk = 0; kk < 2; ++kk) {
    v16bf ap = frag_a(&Pm[0][0], 64, mrow, half, kk * 32);
    v16bf bv = frag_b_kn(&Vv[0][0], DH, lane & 15, half, kk * 32);
    o = wmma_bf16(ap, bv, o);
  }
  const int nfeat = lane & 15;
#pragma unroll
  for (int r = 0; r < 8; ++r) {
    const int m = wid * 16 + r + half * 8;
    AO[((size_t)(m * 512 + b)) * CDIM + h * DH + nfeat] = (bf16)o[r];
  }
}

// ---------------------------------------------------------------------------
// Kernel 4: permute long -> short token order, add positional, bf16 cast.
// ---------------------------------------------------------------------------
__global__ __launch_bounds__(256) void long_to_short_kernel(
    const float* __restrict__ VB, const float* __restrict__ pos_q,
    bf16* __restrict__ Xs, bf16* __restrict__ Vs) {
  const int id = blockIdx.x * 256 + threadIdx.x;   // long-token-major (tok,c)
  const int c = id & 127, tok = id >> 7;
  const int bb = tok & 511, l = tok >> 9;
  const int t = l >> 4, bnh = (l >> 2) & 3, bnw = l & 3;
  const int n = bb >> 8, hb = (bb >> 4) & 15, wb = bb & 15;
  const int lp = (t * 16 + hb) * 16 + wb;
  const int bp = (n * 4 + bnh) * 4 + bnw;
  const size_t o = ((size_t)(lp * 32 + bp)) * CDIM + c;
  const size_t src =
      ((((size_t)n * CDIM + c) * TDIM + t) * HDIM + (bnh * 16 + hb)) * WDIM +
      (bnw * 16 + wb);
  const float vbv = VB[id];
  Xs[o] = (bf16)(vbv + pos_q[src]);
  Vs[o] = (bf16)vbv;
}

// ---------------------------------------------------------------------------
// Kernel 5: short attention, flash-style. Block = (b,h,q-tile of 64 rows),
// streams 16 key chunks of 64 via async-to-LDS DMA + prefetch hints.
// ---------------------------------------------------------------------------
__global__ __launch_bounds__(128) void attn_short_kernel(
    const bf16* __restrict__ QP, const bf16* __restrict__ KP,
    const bf16* __restrict__ VP, bf16* __restrict__ AO) {
  const int blk = blockIdx.x;
  const int qt = blk & 15, bh = blk >> 4, b = bh >> 3, h = bh & 7;

  __shared__ alignas(16) bf16 Qs[64][32];
  __shared__ alignas(16) bf16 Ks[64][32];
  __shared__ alignas(16) bf16 Vv[64][DH];
  __shared__ alignas(16) float Sf[64][64];
  __shared__ alignas(16) bf16 Pm[64][64];
  __shared__ float rmax[64], rsum[64], ralpha[64];

  const int tid = threadIdx.x;
  const int rr = tid >> 1, hs = tid & 1;    // 64 rows x two 16B chunks
  {
    const int l = qt * 64 + rr;
    cp_async_b128(QP + ((size_t)(l * 32 + b)) * CDIM + h * DH + hs * 8,
                  &Qs[rr][hs * 8]);
    const uint4 z4 = make_uint4(0u, 0u, 0u, 0u);
    *(uint4*)&Qs[rr][16 + hs * 8] = z4;   // K-dim pads stay zero
    *(uint4*)&Ks[rr][16 + hs * 8] = z4;
  }
  if (tid < 64) {
    rmax[tid] = -3.0e38f;
    rsum[tid] = 0.0f;
  }
  cp_async_wait();
  __syncthreads();

  const int wid = tid >> 5, lane = tid & 31, half = lane >> 4;
  const int mrow = wid * 16 + (lane & 15);
  const v16bf aq = frag_a(&Qs[0][0], 32, mrow, half, 0);
  v8f acc = {};

  for (int kc = 0; kc < 16; ++kc) {
    // Stage K/V chunk (64 rows x 32B each) via async DMA.
    {
      const size_t gb =
          ((size_t)((kc * 64 + rr) * 32 + b)) * CDIM + h * DH + hs * 8;
      cp_async_b128(KP + gb, &Ks[rr][hs * 8]);
      cp_async_b128(VP + gb, &Vv[rr][hs * 8]);
    }
    // Hint the following chunk toward the caches (global_prefetch_b8).
    if (kc + 1 < 16) {
      const size_t ntok =
          ((size_t)(((kc + 1) * 64 + rr) * 32 + b)) * CDIM + h * DH;
      __builtin_prefetch(KP + ntok, 0, 1);
      __builtin_prefetch(VP + ntok, 0, 1);
    }
    cp_async_wait();
    __syncthreads();

    // S chunk = Q K^T
    for (int nt = 0; nt < 4; ++nt) {
      const int ncol = nt * 16 + (lane & 15);
      v16bf bk = frag_b_nk(&Ks[0][0], 32, ncol, half, 0);
      v8f s = {};
      s = wmma_bf16(aq, bk, s);
#pragma unroll
      for (int r = 0; r < 8; ++r) Sf[wid * 16 + r + half * 8][ncol] = s[r];
    }
    __syncthreads();

    // Online softmax update per row.
    if (tid < 64) {
      float mx = rmax[tid];
      for (int j = 0; j < 64; ++j) mx = fmaxf(mx, Sf[tid][j]);
      const float al = __expf(rmax[tid] - mx);
      float s = 0.0f;
      for (int j = 0; j < 64; ++j) {
        const float p = __expf(Sf[tid][j] - mx);
        Pm[tid][j] = (bf16)p;
        s += p;
      }
      rsum[tid] = rsum[tid] * al + s;
      rmax[tid] = mx;
      ralpha[tid] = al;
    }
    __syncthreads();

    // Rescale accumulator, then acc += P_chunk @ V_chunk (K=64).
#pragma unroll
    for (int r = 0; r < 8; ++r) {
      const float al = ralpha[wid * 16 + r + half * 8];
      acc[r] = acc[r] * al;
    }
#pragma unroll
    for (int kk = 0; kk < 2; ++kk) {
      v16bf ap = frag_a(&Pm[0][0], 64, mrow, half, kk * 32);
      v16bf bv = frag_b_kn(&Vv[0][0], DH, lane & 15, half, kk * 32);
      acc = wmma_bf16(ap, bv, acc);
    }
    __syncthreads();   // Ks/Vv/Sf reused next iteration
  }

  const int nfeat = lane & 15;
#pragma unroll
  for (int r = 0; r < 8; ++r) {
    const int ml = wid * 16 + r + half * 8;
    const int l = qt * 64 + ml;
    const float val = acc[r] / rsum[ml];
    AO[((size_t)(l * 32 + b)) * CDIM + h * DH + nfeat] = (bf16)val;
  }
}

// ---------------------------------------------------------------------------
// Host-side orchestration.
// ---------------------------------------------------------------------------
extern "C" void kernel_launch(void* const* d_in, const int* in_sizes, int n_in,
                              void* d_out, int out_size, void* d_ws,
                              size_t ws_size, hipStream_t stream) {
  (void)in_sizes; (void)n_in; (void)out_size; (void)ws_size;
  const float* q  = (const float*)d_in[0];
  const float* k  = (const float*)d_in[1];
  const float* v  = (const float*)d_in[2];
  const float* pq = (const float*)d_in[3];
  const float* pk = (const float*)d_in[4];
  const float* w_in_long   = (const float*)d_in[5];
  const float* b_in_long   = (const float*)d_in[6];
  const float* w_out_long  = (const float*)d_in[7];
  const float* b_out_long  = (const float*)d_in[8];
  const float* w_in_short  = (const float*)d_in[9];
  const float* b_in_short  = (const float*)d_in[10];
  const float* w_out_short = (const float*)d_in[11];
  const float* b_out_short = (const float*)d_in[12];

  const size_t MC = (size_t)MTOK * CDIM;
  bf16* Xq = (bf16*)d_ws;
  bf16* Xk = Xq + MC;
  bf16* Xv = Xk + MC;
  bf16* QP = Xv + MC;
  bf16* KP = QP + MC;
  bf16* VP = KP + MC;
  bf16* AO = VP + MC;
  float* VB = (float*)(AO + MC);

  const float qscale = 0.25f;   // 1/sqrt(dh), dh=16

  // ---- long stage ----
  pack_long_kernel<<<16384, 256, 0, stream>>>(q, k, v, pq, pk, Xq, Xk, Xv);
  gemm_tc_kernel<0><<<dim3(MTOK / 64, 1, 3), 256, 0, stream>>>(
      Xq, Xk, Xv, w_in_long, b_in_long, QP, KP, VP, nullptr, qscale);
  attn_long_kernel<<<4096, 128, 0, stream>>>(QP, KP, VP, AO);
  gemm_tc_kernel<1><<<dim3(MTOK / 64, 1, 1), 256, 0, stream>>>(
      AO, AO, AO, w_out_long, b_out_long, nullptr, nullptr, nullptr, VB, 1.0f);

  // ---- long -> short permute (+pos) ----
  long_to_short_kernel<<<16384, 256, 0, stream>>>(VB, pq, Xq, Xv);

  // ---- short stage (q = k = vs+pqs, v = vs) ----
  gemm_tc_kernel<0><<<dim3(MTOK / 64, 1, 3), 256, 0, stream>>>(
      Xq, Xq, Xv, w_in_short, b_in_short, QP, KP, VP, nullptr, qscale);
  attn_short_kernel<<<4096, 128, 0, stream>>>(QP, KP, VP, AO);
  gemm_tc_kernel<2><<<dim3(MTOK / 64, 1, 1), 256, 0, stream>>>(
      AO, AO, AO, w_out_short, b_out_short, nullptr, nullptr, nullptr,
      (float*)d_out, 1.0f);
}
